// JointNetwork_48988396978795
// MI455X (gfx1250) — compile-verified, matrix operand order checked by reading, not run on hardware
//
#include <hip/hip_runtime.h>
#include <hip/hip_bf16.h>

typedef __attribute__((ext_vector_type(16))) _Float16 v16h;
typedef __attribute__((ext_vector_type(8)))  _Float16 v8h;
typedef __attribute__((ext_vector_type(2)))  _Float16 v2h;
typedef __attribute__((ext_vector_type(8)))  float    v8f;
typedef __attribute__((ext_vector_type(2)))  float    v2f;

#define BB    4
#define TT    512
#define UU    100
#define AFEAT 512
#define TFEAT 320
#define HID   640
#define NCLS  512

// ---------------------------------------------------------------------------
// gfx1250 hardware tanh (TRANS op). Builtin if available, else inline asm with
// a trailing v_nop to satisfy the 1-instruction TRANS result-use hazard.
// ---------------------------------------------------------------------------
__device__ __forceinline__ float fast_tanh(float x) {
#if __has_builtin(__builtin_amdgcn_tanhf)
  return __builtin_amdgcn_tanhf(x);
#elif __has_builtin(__builtin_amdgcn_tanh_f32)
  return __builtin_amdgcn_tanh_f32(x);
#else
  float r;
  asm volatile("v_tanh_f32 %0, %1\n\tv_nop" : "=v"(r) : "v"(x));
  return r;
#endif
}

// ---------------------------------------------------------------------------
// Kernel 1a: W_j [HID,NCLS] fp32 -> Wjt [NCLS,HID] f16 (transpose + convert).
// ---------------------------------------------------------------------------
__global__ __launch_bounds__(256) void wj_transpose_f16(
    const float* __restrict__ Wj, _Float16* __restrict__ Wjt) {
  int idx = blockIdx.x * 256 + threadIdx.x;       // 0 .. NCLS*HID-1
  int v = idx / HID;
  int k = idx - v * HID;
  Wjt[idx] = (_Float16)Wj[(size_t)k * NCLS + v];
}

// ---------------------------------------------------------------------------
// Kernel 1b: generic fp32 transpose  W[K,N] -> Wt[N,K]  (for W_a, W_t so the
// proj kernel's B-fragments become contiguous 8B loads).
// ---------------------------------------------------------------------------
__global__ __launch_bounds__(256) void w_transpose_f32(
    const float* __restrict__ W, float* __restrict__ Wt, int K, int N) {
  int idx = blockIdx.x * 256 + threadIdx.x;       // 0 .. N*K-1
  int n = idx / K;
  int k = idx - n * K;
  Wt[idx] = W[(size_t)k * N + n];
}

// ---------------------------------------------------------------------------
// Kernel 2: exact fp32 projection  C[M,N] = A[M,K] @ W[K,N] + bias[N]
// using v_wmma_f32_16x16x4_f32 with pre-transposed weights Wt[N,K].
// One 16-row M tile per workgroup, 8 waves x 5 N-tiles cover N = 640.
// K = 16 per iteration: a batched load phase (24 x b64) followed by 20 WMMAs
// so the loads clause together and the waits stagger instead of serializing.
//
// fp32 A 16x4 layout: lane L holds row M=L%16, element i -> K = 2*(L/16)+i.
// B column-major symmetric: lane L holds col N=L%16, same K map -> from
// Wt[N,K] this is one contiguous float2 per (tile, sub-step).
// D: lane L col N=L%16, VGPR r -> row M = r + 8*(L/16).
// ---------------------------------------------------------------------------
__global__ __launch_bounds__(256) void proj_wmma_f32(
    const float* __restrict__ A, const float* __restrict__ Wt,
    const float* __restrict__ bias, float* __restrict__ Cout,
    int M, int K) {
  const int N = HID;                       // 640 = 8 waves * 5 tiles * 16
  int m0   = blockIdx.x * 16;
  int lane = threadIdx.x & 31;
  int wv   = threadIdx.x >> 5;
  int g    = lane >> 4;
  int r    = lane & 15;

  int m     = m0 + r;
  int m_eff = (m < M) ? m : (M - 1);
  const float* arow = A + (size_t)m_eff * K + 2 * g;

  const float* wrow[5];
#pragma unroll
  for (int j = 0; j < 5; ++j)
    wrow[j] = Wt + (size_t)((wv * 5 + j) * 16 + r) * K + 2 * g;

  v8f z = {0.f, 0.f, 0.f, 0.f, 0.f, 0.f, 0.f, 0.f};
  v8f acc[5] = {z, z, z, z, z};

  for (int k = 0; k < K; k += 16) {
    // ---- load phase: everything for this K=16 slab, batched ----
    v2f afr[4];
#pragma unroll
    for (int s = 0; s < 4; ++s) {
      float2 av = *(const float2*)(arow + k + 4 * s);
      afr[s][0] = av.x;
      afr[s][1] = av.y;
    }
    v2f bfr[5][4];
#pragma unroll
    for (int j = 0; j < 5; ++j) {
#pragma unroll
      for (int s = 0; s < 4; ++s) {
        float2 bv = *(const float2*)(wrow[j] + k + 4 * s);
        bfr[j][s][0] = bv.x;
        bfr[j][s][1] = bv.y;
      }
    }
    // ---- compute phase: 20 WMMAs, waits drain progressively ----
#pragma unroll
    for (int s = 0; s < 4; ++s) {
#pragma unroll
      for (int j = 0; j < 5; ++j) {
        acc[j] = __builtin_amdgcn_wmma_f32_16x16x4_f32(
            false, afr[s], false, bfr[j][s], (short)0, acc[j], false, false);
      }
    }
  }

#pragma unroll
  for (int j = 0; j < 5; ++j) {
    int n = (wv * 5 + j) * 16 + r;
    float bb = bias[n];
#pragma unroll
    for (int rr = 0; rr < 8; ++rr) {
      int mr = m0 + 8 * g + rr;
      if (mr < M) Cout[(size_t)mr * N + n] = acc[j][rr] + bb;
    }
  }
}

// ---------------------------------------------------------------------------
// Kernel 3: fused joint  out[b,t,u,v] = tanh(a + t) @ Wjt + b_j
//
// One WG (512 threads = 16 waves) per (b, t, u-half-of-64-rows):
//   - 4 M-tiles of 16 u-rows, staged through LDS (double buffered)
//   - wave w owns columns [w*32, w*32+32) -> 2 N-tiles
//   - K loop: 20 chunks of 32, v_wmma_f32_16x16x32_f16, fp32 accumulation
//
// A-fragment layout in LDS matches the ISA register layout exactly:
//   lane L, half j  ->  row M = L%16, K = (j<8?0:16) + 8*(L/16) + (j&7)
// so each wave fetches a fragment with two ds_load_b128 per tile.
// ---------------------------------------------------------------------------
__global__ __launch_bounds__(512) void joint_wmma_f16(
    const float* __restrict__ a_proj,   // [B*T, HID]
    const float* __restrict__ t_proj,   // [B*U, HID]
    const _Float16* __restrict__ Wjt,   // [NCLS, HID]
    const float* __restrict__ b_j,      // [NCLS]
    float* __restrict__ out) {          // [B, T, U, NCLS]
  __shared__ _Float16 sA[2][4][512];    // [buf][m-tile][lane*16 + half]

  int wg = blockIdx.x;
  int uh = wg & 1;             // u half: rows [uh*64, uh*64+64)
  int bt = wg >> 1;            // b*T + t
  int b  = bt / TT;
  int u_base = uh * 64;
  // number of valid 16-row M tiles in this half (4 for lower, 3 for upper)
  int mtCount = (UU - u_base + 15) >> 4;

  int tid  = threadIdx.x;
  int lane = tid & 31;
  int w    = tid >> 5;         // wave 0..15
  int g    = lane >> 4;
  int r    = lane & 15;

  // ---- store-phase mapping: thread -> (row m of 64, k-pair p2 of 8) ----
  int m_s  = tid >> 3;         // 0..63
  int p2   = tid & 7;          // k pairs {2*p2, 2*p2+1} and {+16}
  int g_s  = (p2 >> 2) & 1;
  int lane_s = (m_s & 15) + 16 * g_s;
  int mt_s   = m_s >> 4;
  int jp     = (p2 & 3) * 2;   // half index within lo (h=0) run

  int u_row_s = u_base + m_s;
  int u_eff   = (u_row_s < UU) ? u_row_s : (UU - 1);

  const float* arow = a_proj + (size_t)bt * HID;
  const float* trow = t_proj + ((size_t)b * UU + u_eff) * HID;

  v8f z = {0.f, 0.f, 0.f, 0.f, 0.f, 0.f, 0.f, 0.f};
  v8f acc[4][2] = {{z, z}, {z, z}, {z, z}, {z, z}};

  for (int kc = 0; kc < HID / 32; ++kc) {
    int k0  = kc * 32;
    int buf = kc & 1;

    // ---- cooperative joint-chunk compute: 64 rows x 32 K, once per WG ----
#pragma unroll
    for (int h = 0; h < 2; ++h) {
      int k = k0 + 2 * p2 + 16 * h;          // pair {k, k+1}
      float2 av = *(const float2*)(arow + k);
      float2 tv = *(const float2*)(trow + k);
      v2h pk;
      pk[0] = (_Float16)fast_tanh(av.x + tv.x);
      pk[1] = (_Float16)fast_tanh(av.y + tv.y);
      *(v2h*)&sA[buf][mt_s][lane_s * 16 + jp + 8 * h] = pk;
    }

    __syncthreads();   // stores of this buf visible; prev-buf reads done

    // ---- B fragments: 2 N-tiles, two contiguous 16B loads each ----
    v16h bf[2];
#pragma unroll
    for (int nt = 0; nt < 2; ++nt) {
      const _Float16* wrow = Wjt + (size_t)(w * 32 + nt * 16 + r) * HID;
      v8h blo = *(const v8h*)(wrow + k0 + 8 * g);
      v8h bhi = *(const v8h*)(wrow + k0 + 16 + 8 * g);
      bf[nt] = __builtin_shufflevector(blo, bhi, 0, 1, 2, 3, 4, 5, 6, 7, 8,
                                       9, 10, 11, 12, 13, 14, 15);
    }

    // ---- A fragments from LDS + 8 WMMAs (uniform tail-tile skip) ----
#pragma unroll
    for (int mt = 0; mt < 4; ++mt) {
      if (mt < mtCount) {                       // block-uniform guard
        const _Float16* sp = &sA[buf][mt][lane * 16];
        v8h alo = *(const v8h*)(sp);
        v8h ahi = *(const v8h*)(sp + 8);
        v16h af = __builtin_shufflevector(alo, ahi, 0, 1, 2, 3, 4, 5, 6, 7,
                                          8, 9, 10, 11, 12, 13, 14, 15);
#pragma unroll
        for (int nt = 0; nt < 2; ++nt) {
          acc[mt][nt] = __builtin_amdgcn_wmma_f32_16x16x32_f16(
              false, af, false, bf[nt], (short)0, acc[mt][nt], false, false);
        }
      }
    }
  }

  // ---- epilogue: D row M = rr + 8*g, col N = r; fuse bias; mask u tail ----
#pragma unroll
  for (int mt = 0; mt < 4; ++mt) {
    if (mt < mtCount) {
#pragma unroll
      for (int nt = 0; nt < 2; ++nt) {
        int v = w * 32 + nt * 16 + r;
        float bj = b_j[v];
        float* ob = out + (size_t)bt * UU * NCLS + v;
#pragma unroll
        for (int rr = 0; rr < 8; ++rr) {
          int u_row = u_base + mt * 16 + 8 * g + rr;
          if (u_row < UU) ob[(size_t)u_row * NCLS] = acc[mt][nt][rr] + bj;
        }
      }
    }
  }
}

// ---------------------------------------------------------------------------
extern "C" void kernel_launch(void* const* d_in, const int* in_sizes, int n_in,
                              void* d_out, int out_size, void* d_ws,
                              size_t ws_size, hipStream_t stream) {
  const float* audio = (const float*)d_in[0];  // [B,T,AFEAT]
  const float* text  = (const float*)d_in[1];  // [B,U,TFEAT]
  const float* W_a   = (const float*)d_in[2];  // [AFEAT,HID]
  const float* b_a   = (const float*)d_in[3];  // [HID]
  const float* W_t   = (const float*)d_in[4];  // [TFEAT,HID]
  const float* b_t   = (const float*)d_in[5];  // [HID]
  const float* W_j   = (const float*)d_in[6];  // [HID,NCLS]
  const float* b_j   = (const float*)d_in[7];  // [NCLS]
  float* out = (float*)d_out;

  // ---- workspace layout (all 16B-aligned) ----
  char* ws = (char*)d_ws;
  size_t off = 0;
  float*    a_proj = (float*)(ws + off);  off += (size_t)BB * TT * HID * 4;   // 5.24 MB
  float*    t_proj = (float*)(ws + off);  off += (size_t)BB * UU * HID * 4;   // 1.02 MB
  _Float16* Wjt    = (_Float16*)(ws + off); off += (size_t)NCLS * HID * 2;    // 0.66 MB
  float*    Wat    = (float*)(ws + off);  off += (size_t)HID * AFEAT * 4;     // 1.31 MB
  float*    Wtt    = (float*)(ws + off);  off += (size_t)HID * TFEAT * 4;     // 0.82 MB

  // 1) pre-transpose weights
  wj_transpose_f16<<<(NCLS * HID) / 256, 256, 0, stream>>>(W_j, Wjt);
  w_transpose_f32<<<(HID * AFEAT) / 256, 256, 0, stream>>>(W_a, Wat, AFEAT,
                                                           HID);
  w_transpose_f32<<<(HID * TFEAT) / 256, 256, 0, stream>>>(W_t, Wtt, TFEAT,
                                                           HID);

  // 2) exact fp32 projections (v_wmma_f32_16x16x4_f32)
  proj_wmma_f32<<<(BB * TT) / 16, 256, 0, stream>>>(audio, Wat, b_a, a_proj,
                                                    BB * TT, AFEAT);
  proj_wmma_f32<<<(BB * UU + 15) / 16, 256, 0, stream>>>(text, Wtt, b_t,
                                                         t_proj, BB * UU,
                                                         TFEAT);

  // 3) fused tanh-joint GEMM (v_wmma_f32_16x16x32_f16 + v_tanh_f32 + LDS)
  joint_wmma_f16<<<BB * TT * 2, 512, 0, stream>>>(a_proj, t_proj, Wjt, b_j,
                                                  out);
}
